// SelfAttention_34016140984727
// MI455X (gfx1250) — matrix operand order repaired and checked by reading, not compiled, not run on hardware
//
#include <hip/hip_runtime.h>
#include <hip/hip_bf16.h>

// ---------------------------------------------------------------------------
// Self-attention forward for MI455X (gfx1250), bf16 WMMA pipeline.
// B=2, S=2048, EMB=1024, H=16, D=64.
// ---------------------------------------------------------------------------

typedef __attribute__((ext_vector_type(16))) __bf16 v16bf;
typedef __attribute__((ext_vector_type(8)))  __bf16 v8bf;
typedef __attribute__((ext_vector_type(4)))  __bf16 v4bf;
typedef __attribute__((ext_vector_type(8)))  float  v8f;

#define WMMA_BF16(a, b, c) \
  __builtin_amdgcn_wmma_f32_16x16x32_bf16(false, (a), false, (b), (short)0, (c), false, false)

constexpr int kB   = 2;
constexpr int kS   = 2048;
constexpr int kEMB = 1024;
constexpr int kH   = 16;
constexpr int kD   = 64;
constexpr int kBS  = kB * kS;          // 4096 rows total
constexpr float kSCALE = 0.125f;       // 1/sqrt(64), folded into Q projection

// ---------------------------------------------------------------------------
// fp32 -> bf16 conversion (4-wide)
// ---------------------------------------------------------------------------
__global__ void cvt_f32_bf16(const float* __restrict__ src,
                             __bf16* __restrict__ dst, int n4) {
  int i = blockIdx.x * blockDim.x + threadIdx.x;
  if (i >= n4) return;
  float4 v = ((const float4*)src)[i];
  v4bf o;
  o[0] = (__bf16)v.x; o[1] = (__bf16)v.y; o[2] = (__bf16)v.z; o[3] = (__bf16)v.w;
  ((v4bf*)dst)[i] = o;
}

// ---------------------------------------------------------------------------
// GEMM: out = (X(bf16)[kBS,kEMB] @ W(bf16)[kEMB,kEMB]^T + bias(f32)) * oscale
// One wave computes a 32x64 output tile: 2 A-frags share 4 B-frags per k-step
// (8 WMMAs / 6 x 32B loads per lane per k-step).
// grid = (kBS/32, kEMB/64/4), block = 128 (4 waves, one col-group each).
// MODE 0: write bf16 head-split [B,H,S,D]             (Q, K)
// MODE 1: write bf16 head-split transposed [B,H,D,S]  (V)
// MODE 2: write f32 row-major [kBS,kEMB]              (final output projection)
// ---------------------------------------------------------------------------
template <int MODE>
__global__ __launch_bounds__(128) void proj_kernel(
    const __bf16* __restrict__ X, const __bf16* __restrict__ W,
    const float* __restrict__ bias, void* __restrict__ outv, float oscale) {
  const int lane = threadIdx.x & 31;
  const int wave = threadIdx.x >> 5;
  const int row0 = blockIdx.x * 32;
  const int col0 = (blockIdx.y * 4 + wave) * 64;

  const int m    = lane & 15;        // row within tile (A) / col within tile (B)
  const int half = lane >> 4;
  const int kb   = half * 8;         // unused; A and B both load at kb16
  const int kb16 = half * 16;        // B fragment K sub-offset

  v8f acc[2][4];
#pragma unroll
  for (int j = 0; j < 2; ++j)
#pragma unroll
    for (int t = 0; t < 4; ++t) acc[j][t] = v8f{};

  const __bf16* arow0 = X + (size_t)(row0 + m) * kEMB;
  const __bf16* arow1 = X + (size_t)(row0 + 16 + m) * kEMB;
  const __bf16* brow[4];
#pragma unroll
  for (int t = 0; t < 4; ++t)
    brow[t] = W + (size_t)(col0 + t * 16 + m) * kEMB + kb16;

  for (int kk = 0; kk < kEMB; kk += 32) {
    // prefetch next k-chunk (lowers to global_prefetch_b8)
    if (kk + 256 < kEMB) {
      __builtin_prefetch(arow0 + kk + 256, 0, 0);
      __builtin_prefetch(arow1 + kk + 256, 0, 0);
      __builtin_prefetch(brow[0] + kk + 256, 0, 0);
    }
    // Two A fragments (rows row0..+15 and row0+16..+31)
    v16bf a0, a1;
    {
      v8bf lo = *(const v8bf*)(arow0 + kk + kb16);
      v8bf hi = *(const v8bf*)(arow0 + kk + kb16 + 8);
#pragma unroll
      for (int i = 0; i < 8; ++i) { a0[i] = lo[i]; a0[i + 8] = hi[i]; }
      lo = *(const v8bf*)(arow1 + kk + kb16);
      hi = *(const v8bf*)(arow1 + kk + kb16 + 8);
#pragma unroll
      for (int i = 0; i < 8; ++i) { a1[i] = lo[i]; a1[i + 8] = hi[i]; }
    }
    // 4 B fragments (32x16 each), shared by both A fragments
#pragma unroll
    for (int t = 0; t < 4; ++t) {
      v16bf bf = *(const v16bf*)(brow[t] + kk);
      acc[0][t] = WMMA_BF16(a0, bf, acc[0][t]);
      acc[1][t] = WMMA_BF16(a1, bf, acc[1][t]);
    }
  }

  // Epilogue: C layout — lane holds col n = col0+t*16+m, rows half*8 + r.
#pragma unroll
  for (int j = 0; j < 2; ++j) {
    const int rbase = j * 16 + half * 8;
#pragma unroll
    for (int t = 0; t < 4; ++t) {
      const int n  = col0 + t * 16 + m;
      const float bn = bias[n];
      if constexpr (MODE == 2) {
        float* out = (float*)outv;
#pragma unroll
        for (int r = 0; r < 8; ++r) {
          const int row = row0 + rbase + r;
          out[(size_t)row * kEMB + n] = (acc[j][t][r] + bn) * oscale;
        }
      } else {
        __bf16* out = (__bf16*)outv;
        const int h = n >> 6;          // head index
        const int d = n & 63;          // dim within head
        if constexpr (MODE == 0) {     // [B,H,S,D]
#pragma unroll
          for (int r = 0; r < 8; ++r) {
            const int row = row0 + rbase + r;
            const int bb = row >> 11;             // row / kS
            const int s  = row & (kS - 1);
            out[(((size_t)bb * kH + h) * kS + s) * kD + d] =
                (__bf16)((acc[j][t][r] + bn) * oscale);
          }
        } else {                       // MODE 1: [B,H,D,S] (contiguous 8-wide store)
          const int row = row0 + rbase;
          const int bb = row >> 11;
          const int s0 = row & (kS - 1);
          v8bf tv;
#pragma unroll
          for (int r = 0; r < 8; ++r) tv[r] = (__bf16)((acc[j][t][r] + bn) * oscale);
          *(v8bf*)(out + (((size_t)bb * kH + h) * kD + d) * kS + s0) = tv;
        }
      }
    }
  }
}

// ---------------------------------------------------------------------------
// Flash attention: one wave per (b, h, 16-query tile).
// Q (pre-scaled by 1/sqrt(D)), K: [B,H,S,D] bf16. Vt: [B,H,D,S] bf16.
// Out: [B,S,EMB] bf16 (heads merged).
// grid = (kS/16/4, kH, kB), block = 128 (4 waves).
// ---------------------------------------------------------------------------
__global__ __launch_bounds__(128) void attn_kernel(
    const __bf16* __restrict__ Q, const __bf16* __restrict__ K,
    const __bf16* __restrict__ Vt, __bf16* __restrict__ Out) {
  __shared__ __align__(64) __bf16 lds[4][16][32];   // per-wave P transpose tile

  const int lane = threadIdx.x & 31;
  const int wave = threadIdx.x >> 5;
  const int qt = blockIdx.x * 4 + wave;   // 16-row query tile
  const int h  = blockIdx.y;
  const int b  = blockIdx.z;

  const int m    = lane & 15;
  const int half = lane >> 4;
  const int kb   = half * 8;
  const int kb16 = half * 16;

  const size_t head = ((size_t)b * kH + h) * (size_t)kS * kD;

  // Q A-fragments for d-chunks {0..31}, {32..63}
  const __bf16* qrow = Q + head + (size_t)(qt * 16 + m) * kD;
  v16bf aq[2];
#pragma unroll
  for (int c = 0; c < 2; ++c) {
    v8bf lo = *(const v8bf*)(qrow + c * 32 + kb16);
    v8bf hi = *(const v8bf*)(qrow + c * 32 + kb16 + 8);
#pragma unroll
    for (int i = 0; i < 8; ++i) { aq[c][i] = lo[i]; aq[c][i + 8] = hi[i]; }
  }

  float mrow[8], lrow[8];
  v8f o[4];
#pragma unroll
  for (int r = 0; r < 8; ++r) { mrow[r] = -3.0e38f; lrow[r] = 0.0f; }
#pragma unroll
  for (int t = 0; t < 4; ++t) o[t] = v8f{};

  const __bf16* Kb = K + head;
  const __bf16* Vb = Vt + head;

  const int ntiles = ((qt * 16 + 15) >> 5) + 1;   // causal: 32-key chunks needed
  for (int kt = 0; kt < ntiles; ++kt) {
    const int key0 = kt * 32;

    // --- scores: S = Q * K^T, two 16x16 tiles (keys 0..15 / 16..31) ---
    v8f s0 = v8f{}, s1 = v8f{};
#pragma unroll
    for (int c = 0; c < 2; ++c) {
      const int kk = c * 32;
      v16bf b0 = *(const v16bf*)(Kb + (size_t)(key0 + m) * kD + kk + kb16);
      v16bf b1 = *(const v16bf*)(Kb + (size_t)(key0 + 16 + m) * kD + kk + kb16);
      s0 = WMMA_BF16(aq[c], b0, s0);
      s1 = WMMA_BF16(aq[c], b1, s1);
    }

    // --- online softmax (row stats across 16-lane groups) ---
    float alpha[8];
#pragma unroll
    for (int r = 0; r < 8; ++r) {
      const int qpos = qt * 16 + half * 8 + r;
      float x0 = s0[r];
      float x1 = s1[r];
      if (key0 + m > qpos)      x0 = -1.0e30f;   // causal mask
      if (key0 + 16 + m > qpos) x1 = -1.0e30f;
      float vmax = fmaxf(x0, x1);
#pragma unroll
      for (int off = 1; off < 16; off <<= 1)
        vmax = fmaxf(vmax, __shfl_xor(vmax, off, 32));
      const float mnew = fmaxf(mrow[r], vmax);
      alpha[r] = __expf(mrow[r] - mnew);
      mrow[r] = mnew;
      const float p0 = __expf(x0 - mnew);
      const float p1 = __expf(x1 - mnew);
      float ps = p0 + p1;
#pragma unroll
      for (int off = 1; off < 16; off <<= 1)
        ps += __shfl_xor(ps, off, 32);
      lrow[r] = lrow[r] * alpha[r] + ps;
      // stage P into LDS (C-layout -> memory [row][key]) for A-frag reload
      lds[wave][half * 8 + r][m]      = (__bf16)p0;
      lds[wave][half * 8 + r][16 + m] = (__bf16)p1;
    }

    // rescale O accumulators (row replication pattern matches C layout)
#pragma unroll
    for (int t = 0; t < 4; ++t)
#pragma unroll
      for (int r = 0; r < 8; ++r) o[t][r] *= alpha[r];

    // --- reload P as A-fragment (16x32) from LDS (DS ops in-order per wave) ---
    v16bf pfrag;
    {
      v8bf lo = *(const v8bf*)(&lds[wave][m][kb16]);
      v8bf hi = *(const v8bf*)(&lds[wave][m][kb16 + 8]);
#pragma unroll
      for (int i = 0; i < 8; ++i) { pfrag[i] = lo[i]; pfrag[i + 8] = hi[i]; }
    }

    // --- O += P * V : 4 output d-tiles, B-frags from transposed V ---
#pragma unroll
    for (int t = 0; t < 4; ++t) {
      v16bf vf = *(const v16bf*)(Vb + (size_t)(t * 16 + m) * kS + key0 + kb16);
      o[t] = WMMA_BF16(pfrag, vf, o[t]);
    }
  }

  // --- finalize: O /= l, write bf16 merged-head [B,S,EMB] ---
#pragma unroll
  for (int t = 0; t < 4; ++t) {
#pragma unroll
    for (int r = 0; r < 8; ++r) {
      const int s = qt * 16 + half * 8 + r;
      Out[((size_t)b * kS + s) * kEMB + h * kD + t * 16 + m] =
          (__bf16)(o[t][r] / lrow[r]);
    }
  }
}

// ---------------------------------------------------------------------------
// Launcher
// ---------------------------------------------------------------------------
extern "C" void kernel_launch(void* const* d_in, const int* in_sizes, int n_in,
                              void* d_out, int out_size, void* d_ws, size_t ws_size,
                              hipStream_t stream) {
  // Input order: xq, xk, xv, mask, Wq, bq, Wk, bk, Wv, bv, Wo, bo
  const float* xq = (const float*)d_in[0];
  const float* xk = (const float*)d_in[1];
  const float* xv = (const float*)d_in[2];
  // d_in[3] = mask (tril) — causality is computed analytically in attn_kernel.
  const float* Wq = (const float*)d_in[4];
  const float* bq = (const float*)d_in[5];
  const float* Wk = (const float*)d_in[6];
  const float* bk = (const float*)d_in[7];
  const float* Wv = (const float*)d_in[8];
  const float* bv = (const float*)d_in[9];
  const float* Wo = (const float*)d_in[10];
  const float* bo = (const float*)d_in[11];

  constexpr size_t NX = (size_t)kBS * kEMB;        // 4M elems
  constexpr size_t NW = (size_t)kEMB * kEMB;       // 1M elems
  char* ws = (char*)d_ws;
  size_t off = 0;
  auto alloc = [&](size_t elems) {
    __bf16* p = (__bf16*)(ws + off);
    off += ((elems * sizeof(__bf16) + 255) & ~(size_t)255);
    return p;
  };
  __bf16* xq_bf = alloc(NX);
  __bf16* xk_bf = alloc(NX);
  __bf16* xv_bf = alloc(NX);
  __bf16* Wq_bf = alloc(NW);
  __bf16* Wk_bf = alloc(NW);
  __bf16* Wv_bf = alloc(NW);
  __bf16* Wo_bf = alloc(NW);
  __bf16* Q_bf  = alloc(NX);   // [B,H,S,D], pre-scaled by 1/sqrt(D)
  __bf16* K_bf  = alloc(NX);   // [B,H,S,D]
  __bf16* Vt_bf = alloc(NX);   // [B,H,D,S]
  __bf16* A_bf  = alloc(NX);   // attention output, [B,S,EMB]
  (void)ws_size; (void)in_sizes; (void)n_in; (void)out_size;

  // 1) convert activations + weights to bf16
  const int nx4 = (int)(NX / 4), nw4 = (int)(NW / 4);
  cvt_f32_bf16<<<dim3((nx4 + 255) / 256), 256, 0, stream>>>(xq, xq_bf, nx4);
  cvt_f32_bf16<<<dim3((nx4 + 255) / 256), 256, 0, stream>>>(xk, xk_bf, nx4);
  cvt_f32_bf16<<<dim3((nx4 + 255) / 256), 256, 0, stream>>>(xv, xv_bf, nx4);
  cvt_f32_bf16<<<dim3((nw4 + 255) / 256), 256, 0, stream>>>(Wq, Wq_bf, nw4);
  cvt_f32_bf16<<<dim3((nw4 + 255) / 256), 256, 0, stream>>>(Wk, Wk_bf, nw4);
  cvt_f32_bf16<<<dim3((nw4 + 255) / 256), 256, 0, stream>>>(Wv, Wv_bf, nw4);
  cvt_f32_bf16<<<dim3((nw4 + 255) / 256), 256, 0, stream>>>(Wo, Wo_bf, nw4);

  // 2) QKV projections (WMMA GEMMs); 1/sqrt(D) folded into Q
  dim3 pgrid(kBS / 32, kEMB / 64 / 4);
  proj_kernel<0><<<pgrid, 128, 0, stream>>>(xq_bf, Wq_bf, bq, Q_bf, kSCALE);
  proj_kernel<0><<<pgrid, 128, 0, stream>>>(xk_bf, Wk_bf, bk, K_bf, 1.0f);
  proj_kernel<1><<<pgrid, 128, 0, stream>>>(xv_bf, Wv_bf, bv, Vt_bf, 1.0f);

  // 3) flash attention
  attn_kernel<<<dim3(kS / 16 / 4, kH, kB), 128, 0, stream>>>(Q_bf, K_bf, Vt_bf, A_bf);

  // 4) output projection -> f32 d_out
  proj_kernel<2><<<pgrid, 128, 0, stream>>>(A_bf, Wo_bf, bo, d_out, 1.0f);
}
